// BasicTransformerBlock_85581518340362
// MI455X (gfx1250) — matrix operand order repaired and checked by reading, not compiled, not run on hardware
//
#include <hip/hip_runtime.h>
#include <hip/hip_bf16.h>
#include <stdint.h>

// ---------------------------------------------------------------------------
// Types matching gfx1250 WMMA builtin signatures
// ---------------------------------------------------------------------------
typedef __attribute__((ext_vector_type(16))) __bf16 v16bf;
typedef __attribute__((ext_vector_type(8)))  __bf16 v8bf;
typedef __attribute__((ext_vector_type(8)))  float  v8f;

#define D_MODEL 1024
#define N_TOK   2048
#define BATCH   2
#define TOKENS  (BATCH * N_TOK)      // 4096
#define NHEAD   16
#define DHEAD   64
#define IDIM    (NHEAD * DHEAD)      // 1024
#define DCTX    768
#define MCTX    77
#define MPAD    96                   // padded cross-attn key count (3 x 32)
#define FFDIM   4096
#define SCALE_F 0.125f               // DH^-0.5

// ---------------------------------------------------------------------------
// WMMA fragment helpers (contiguous k order used by this file, wave32)
//   A 16x32 bf16 : lane half=lane>>4, row m=lane&15
//                  elem e -> K = half*16 + e
//   B 32x16 bf16 : col n = lane&15, elem e -> K = half*16 + e
//   C/D 16x16 f32: elem e -> row = half*8 + e, col = lane&15
// ---------------------------------------------------------------------------
__device__ __forceinline__ v8bf ld8(const __bf16* p) { return *(const v8bf*)p; }

__device__ __forceinline__ v16bf combine16(v8bf lo, v8bf hi) {
  v16bf r;
#pragma unroll
  for (int i = 0; i < 8; ++i) { r[i] = lo[i]; r[i + 8] = hi[i]; }
  return r;
}

// rowp points at (row m, k0) of a row-major (k-contiguous) 32-wide window
__device__ __forceinline__ v16bf load_a_frag(const __bf16* rowp, int half) {
  return combine16(ld8(rowp + half * 16), ld8(rowp + half * 16 + 8));
}
// colp points at (col n, k0) of a k-contiguous column store (32-wide window)
__device__ __forceinline__ v16bf load_b_frag(const __bf16* colp, int half) {
  return combine16(ld8(colp + half * 16), ld8(colp + half * 16 + 8));
}

__device__ __forceinline__ v8f wmma_bf16(v16bf a, v16bf b, v8f c) {
  return __builtin_amdgcn_wmma_f32_16x16x32_bf16(false, a, false, b,
                                                 (short)0, c, false, false);
}

// ---------------------------------------------------------------------------
// f32 -> bf16 convert (weights, etc.)
// ---------------------------------------------------------------------------
__global__ void cvt_bf16_kernel(const float* __restrict__ a,
                                __bf16* __restrict__ o, size_t n) {
  size_t stride = (size_t)gridDim.x * blockDim.x;
  for (size_t i = (size_t)blockIdx.x * blockDim.x + threadIdx.x; i < n; i += stride)
    o[i] = (__bf16)a[i];
}

// ---------------------------------------------------------------------------
// LayerNorm: one 256-thread block per row of 1024, bf16 output
// ---------------------------------------------------------------------------
__global__ __launch_bounds__(256)
void ln_kernel(const float* __restrict__ x, const float* __restrict__ g,
               const float* __restrict__ b, __bf16* __restrict__ y) {
  __shared__ float red[8];
  const int row = blockIdx.x, tid = threadIdx.x;
  const float* xr = x + (size_t)row * D_MODEL;
  float v[4];
#pragma unroll
  for (int i = 0; i < 4; ++i) v[i] = xr[tid + i * 256];
  float s = v[0] + v[1] + v[2] + v[3];
#pragma unroll
  for (int o = 16; o > 0; o >>= 1) s += __shfl_xor(s, o, 32);
  if ((tid & 31) == 0) red[tid >> 5] = s;
  __syncthreads();
  float mean = (red[0] + red[1] + red[2] + red[3] +
                red[4] + red[5] + red[6] + red[7]) * (1.0f / D_MODEL);
  __syncthreads();
  float s2 = 0.f;
#pragma unroll
  for (int i = 0; i < 4; ++i) { float d = v[i] - mean; s2 += d * d; }
#pragma unroll
  for (int o = 16; o > 0; o >>= 1) s2 += __shfl_xor(s2, o, 32);
  if ((tid & 31) == 0) red[tid >> 5] = s2;
  __syncthreads();
  float var = (red[0] + red[1] + red[2] + red[3] +
               red[4] + red[5] + red[6] + red[7]) * (1.0f / D_MODEL);
  float rinv = rsqrtf(var + 1e-5f);
#pragma unroll
  for (int i = 0; i < 4; ++i) {
    int c = tid + i * 256;
    y[(size_t)row * D_MODEL + c] = (__bf16)((v[i] - mean) * rinv * g[c] + b[c]);
  }
}

// ---------------------------------------------------------------------------
// GEMM: C[M,N] = A[M,K] @ B[K,N]  (A,B bf16 row-major, f32 accumulate)
//   EPI 0: store bf16
//   EPI 1: store bf16, += bias[n]
//   EPI 2: store f32,  += bias[n] + resid[m,n]
// Block: 256 threads (8 waves), tile 64(M) x 128(N), K-step 32.
// ---------------------------------------------------------------------------
template <int EPI>
__global__ __launch_bounds__(256)
void gemm_bf16_wmma(const __bf16* __restrict__ A, const __bf16* __restrict__ B,
                    const float* __restrict__ bias, const float* __restrict__ resid,
                    void* __restrict__ Cout, int M, int N, int K) {
  __shared__ __bf16 As[64][32];    // row-major (k-contiguous)
  __shared__ __bf16 Bs[128][32];   // transposed: [n][k] (k-contiguous)
  const int tid  = threadIdx.x;
  const int lane = tid & 31, w = tid >> 5;
  const int half = lane >> 4, lm = lane & 15;
  const int wm = w >> 2, wn = w & 3;           // 2 x 4 wave grid
  const int mblk = blockIdx.y * 64, nblk = blockIdx.x * 128;

  v8f acc[2][2] = {};

  const int arow = tid >> 2, akc = (tid & 3) * 8;   // A: 64x32, 8 elems/thread
  const int bkr  = tid >> 3, bnc = (tid & 7) * 16;  // B: 32x128, 16 elems/thread

  for (int kt = 0; kt < K; kt += 32) {
    v8bf va = ld8(A + (size_t)(mblk + arow) * K + kt + akc);
    const __bf16* bp = B + (size_t)(kt + bkr) * N + nblk + bnc;
    v8bf vb0 = ld8(bp), vb1 = ld8(bp + 8);
    if (kt + 32 < K) {
      __builtin_prefetch(A + (size_t)(mblk + arow) * K + kt + 32 + akc, 0, 1);
      __builtin_prefetch(B + (size_t)(kt + 32 + bkr) * N + nblk + bnc, 0, 1);
    }
    *(v8bf*)&As[arow][akc] = va;
#pragma unroll
    for (int i = 0; i < 8; ++i) {
      Bs[bnc + i][bkr]     = vb0[i];
      Bs[bnc + 8 + i][bkr] = vb1[i];
    }
    __syncthreads();

    v16bf af[2], bfg[2];
#pragma unroll
    for (int s = 0; s < 2; ++s) {
      af[s]  = load_a_frag(&As[wm * 32 + s * 16 + lm][0], half);
      bfg[s] = load_b_frag(&Bs[wn * 32 + s * 16 + lm][0], half);
    }
#pragma unroll
    for (int si = 0; si < 2; ++si)
#pragma unroll
      for (int sj = 0; sj < 2; ++sj)
        acc[si][sj] = wmma_bf16(af[si], bfg[sj], acc[si][sj]);
    __syncthreads();
  }

#pragma unroll
  for (int si = 0; si < 2; ++si) {
#pragma unroll
    for (int sj = 0; sj < 2; ++sj) {
      const int r0 = mblk + wm * 32 + si * 16 + half * 8;
      const int c  = nblk + wn * 32 + sj * 16 + lm;
      const float bv = (EPI >= 1) ? bias[c] : 0.f;
#pragma unroll
      for (int e = 0; e < 8; ++e) {
        const size_t off = (size_t)(r0 + e) * N + c;
        float val = acc[si][sj][e] + bv;
        if (EPI == 2) ((float*)Cout)[off] = val + resid[off];
        else          ((__bf16*)Cout)[off] = (__bf16)val;
      }
    }
  }
}

// ---------------------------------------------------------------------------
// Flash-style attention. 1 wave per (16-query tile, head, batch).
// Softmax bookkeeping is done in a row-per-lane layout (lane lm owns query
// row lm; the two 16-lane halves split the 32 keys) so row reductions are
// in-register + a single shfl_xor(16); alpha / 1/l are broadcast back to the
// WMMA C-layout through a 16-float LDS array (no bpermute chains).
//   q   : [b*Nq + n][IDIM] bf16   (cols h*64+d)
//   k   : [b*Mpad + j][IDIM] bf16
//   vT  : [((b*H + h)*64 + d) * Mpad + j] bf16  (key-contiguous)
//   out : [b*Nq + n][IDIM] bf16
// ---------------------------------------------------------------------------
__global__ __launch_bounds__(32)
void attn_kernel(const __bf16* __restrict__ q, const __bf16* __restrict__ k,
                 const __bf16* __restrict__ vT, __bf16* __restrict__ out,
                 int Nq, int Mkeys, int Mpad, float scale) {
  __shared__ float  sst[16][32];   // S tile staging (row-major f32, 2 KB)
  __shared__ __bf16 pls[16][32];   // P tile (row-major bf16, 1 KB)
  __shared__ float  rowmeta[16];   // alpha / 1-over-l broadcast
  const int lane = threadIdx.x;
  const int half = lane >> 4, lm = lane & 15;
  const int n0 = blockIdx.x * 16;
  const int h  = blockIdx.y;
  const int b  = blockIdx.z;

  const __bf16* qrow = q + ((size_t)(b * Nq + n0 + lm)) * IDIM + h * DHEAD;
  v16bf qa0 = load_a_frag(qrow, half);        // d = 0..31
  v16bf qa1 = load_a_frag(qrow + 32, half);   // d = 32..63

  float mrow = -1e30f, lrow = 0.f;            // per query row (dup per half)
  v8f acc[4] = {};

  for (int j0 = 0; j0 < Mpad; j0 += 32) {
    // ---- S = Q K^T via WMMA (C-layout, unscaled), spill to LDS -------------
#pragma unroll
    for (int sub = 0; sub < 2; ++sub) {
      const int key = j0 + sub * 16 + lm;     // this lane's key column
      const __bf16* krow = k + ((size_t)(b * Mpad + key)) * IDIM + h * DHEAD;
      v8f s = {};
      s = wmma_bf16(qa0, load_b_frag(krow, half), s);       // d = 0..31
      s = wmma_bf16(qa1, load_b_frag(krow + 32, half), s);  // d = 32..63
#pragma unroll
      for (int e = 0; e < 8; ++e) sst[half * 8 + e][sub * 16 + lm] = s[e];
    }
    __syncthreads();

    // ---- row-per-lane softmax update ------------------------------------
    float sv[16];
    {
      const float4* rp = (const float4*)&sst[lm][half * 16];
#pragma unroll
      for (int i = 0; i < 4; ++i) {
        float4 t = rp[i];
        sv[4 * i + 0] = t.x; sv[4 * i + 1] = t.y;
        sv[4 * i + 2] = t.z; sv[4 * i + 3] = t.w;
      }
    }
    float rmax = -1e30f;
#pragma unroll
    for (int i = 0; i < 16; ++i) {
      const int key = j0 + half * 16 + i;
      sv[i] = (key < Mkeys) ? sv[i] * scale : -1e30f;
      rmax = fmaxf(rmax, sv[i]);
    }
    rmax = fmaxf(rmax, __shfl_xor(rmax, 16, 32));   // combine the two halves
    const float mnew  = fmaxf(mrow, rmax);
    const float alpha = expf(mrow - mnew);
    mrow = mnew;

    float rs = 0.f;
    v8bf p0, p1;
#pragma unroll
    for (int i = 0; i < 16; ++i) {
      float pv = (sv[i] <= -1e29f) ? 0.f : expf(sv[i] - mnew);
      rs += pv;
      if (i < 8) p0[i] = (__bf16)pv;
      else       p1[i - 8] = (__bf16)pv;
    }
    rs += __shfl_xor(rs, 16, 32);
    lrow = lrow * alpha + rs;

    *(v8bf*)&pls[lm][half * 16]     = p0;   // P already key-contiguous
    *(v8bf*)&pls[lm][half * 16 + 8] = p1;
    rowmeta[lm] = alpha;                    // lanes lm, lm+16: identical value
    __syncthreads();

    // ---- rescale acc (C-layout) and accumulate P x V ---------------------
    float a8[8];
    {
      const float4* ap = (const float4*)&rowmeta[half * 8];
      float4 a0 = ap[0], a1 = ap[1];
      a8[0] = a0.x; a8[1] = a0.y; a8[2] = a0.z; a8[3] = a0.w;
      a8[4] = a1.x; a8[5] = a1.y; a8[6] = a1.z; a8[7] = a1.w;
    }
#pragma unroll
    for (int dt = 0; dt < 4; ++dt)
#pragma unroll
      for (int e = 0; e < 8; ++e) acc[dt][e] *= a8[e];

    v16bf pa = load_a_frag(&pls[lm][0], half);   // P as A 16x32
#pragma unroll
    for (int dt = 0; dt < 4; ++dt) {
      const __bf16* vcol =
          vT + ((size_t)((b * NHEAD + h) * DHEAD + dt * 16 + lm)) * Mpad + j0;
      acc[dt] = wmma_bf16(pa, load_b_frag(vcol, half), acc[dt]);
    }
    __syncthreads();
  }

  // ---- epilogue: divide by l, broadcast 1/l through LDS ------------------
  rowmeta[lm] = 1.f / lrow;
  __syncthreads();
  float i8[8];
  {
    const float4* ip = (const float4*)&rowmeta[half * 8];
    float4 a0 = ip[0], a1 = ip[1];
    i8[0] = a0.x; i8[1] = a0.y; i8[2] = a0.z; i8[3] = a0.w;
    i8[4] = a1.x; i8[5] = a1.y; i8[6] = a1.z; i8[7] = a1.w;
  }
#pragma unroll
  for (int e = 0; e < 8; ++e) {
    const int tok = b * Nq + n0 + half * 8 + e;
#pragma unroll
    for (int dt = 0; dt < 4; ++dt)
      out[(size_t)tok * IDIM + h * DHEAD + dt * 16 + lm] =
          (__bf16)(acc[dt][e] * i8[e]);
  }
}

// ---------------------------------------------------------------------------
// V transpose: vT[((b*H+h)*64+d)*Mpad + j] = v[(b*Mpad+j)*IDIM + h*64 + d]
// ---------------------------------------------------------------------------
__global__ void vtrans_kernel(const __bf16* __restrict__ v,
                              __bf16* __restrict__ vT, int Mpad) {
  const size_t total = (size_t)BATCH * NHEAD * DHEAD * Mpad;
  const size_t stride = (size_t)gridDim.x * blockDim.x;
  for (size_t i = (size_t)blockIdx.x * blockDim.x + threadIdx.x; i < total; i += stride) {
    int j = (int)(i % Mpad); size_t r = i / Mpad;
    int d = (int)(r % DHEAD); r /= DHEAD;
    int h = (int)(r % NHEAD); int bb = (int)(r / NHEAD);
    vT[i] = v[((size_t)(bb * Mpad + j)) * IDIM + h * DHEAD + d];
  }
}

// ---------------------------------------------------------------------------
// Context -> bf16 with zero padding to MPAD rows per batch
// ---------------------------------------------------------------------------
__global__ void ctxpad_kernel(const float* __restrict__ ctx,
                              __bf16* __restrict__ o) {
  const size_t total = (size_t)BATCH * MPAD * DCTX;
  const size_t stride = (size_t)gridDim.x * blockDim.x;
  for (size_t i = (size_t)blockIdx.x * blockDim.x + threadIdx.x; i < total; i += stride) {
    int c = (int)(i % DCTX); size_t r = i / DCTX;
    int bb = (int)(r / MPAD), j = (int)(r % MPAD);
    o[i] = (__bf16)((j < MCTX) ? ctx[((size_t)(bb * MCTX + j)) * DCTX + c] : 0.f);
  }
}

// ---------------------------------------------------------------------------
// GEGLU: out[r, c] = hg[r, c] * gelu(hg[r, FF + c])  (exact erf gelu)
// ---------------------------------------------------------------------------
__global__ void geglu_kernel(const __bf16* __restrict__ hg,
                             __bf16* __restrict__ o) {
  const size_t total = (size_t)TOKENS * FFDIM;
  const size_t stride = (size_t)gridDim.x * blockDim.x;
  for (size_t i = (size_t)blockIdx.x * blockDim.x + threadIdx.x; i < total; i += stride) {
    int c = (int)(i & (FFDIM - 1)); size_t r = i >> 12;
    float hv = (float)hg[r * (2 * FFDIM) + c];
    float gv = (float)hg[r * (2 * FFDIM) + FFDIM + c];
    float ge = 0.5f * gv * (1.f + erff(gv * 0.70710678118654752f));
    o[i] = (__bf16)(hv * ge);
  }
}

// ---------------------------------------------------------------------------
// Orchestration
// ---------------------------------------------------------------------------
static inline size_t align256(size_t x) { return (x + 255) & ~(size_t)255; }

extern "C" void kernel_launch(void* const* d_in, const int* in_sizes, int n_in,
                              void* d_out, int out_size, void* d_ws, size_t ws_size,
                              hipStream_t stream) {
  (void)in_sizes; (void)n_in; (void)out_size; (void)ws_size;
  const float* x    = (const float*)d_in[0];
  const float* ctx  = (const float*)d_in[1];
  const float* ln1g = (const float*)d_in[2];
  const float* ln1b = (const float*)d_in[3];
  const float* ln2g = (const float*)d_in[4];
  const float* ln2b = (const float*)d_in[5];
  const float* ln3g = (const float*)d_in[6];
  const float* ln3b = (const float*)d_in[7];
  const float* Wq1  = (const float*)d_in[8];
  const float* Wk1  = (const float*)d_in[9];
  const float* Wv1  = (const float*)d_in[10];
  const float* Wo1  = (const float*)d_in[11];
  const float* bo1  = (const float*)d_in[12];
  const float* Wq2  = (const float*)d_in[13];
  const float* Wk2  = (const float*)d_in[14];
  const float* Wv2  = (const float*)d_in[15];
  const float* Wo2  = (const float*)d_in[16];
  const float* bo2  = (const float*)d_in[17];
  const float* Wff1 = (const float*)d_in[18];
  const float* bff1 = (const float*)d_in[19];
  const float* Wff2 = (const float*)d_in[20];
  const float* bff2 = (const float*)d_in[21];

  uint8_t* p = (uint8_t*)d_ws;
  auto carve = [&](size_t bytes) -> uint8_t* {
    uint8_t* r = p; p += align256(bytes); return r;
  };
  __bf16* Wq1b  = (__bf16*)carve((size_t)D_MODEL * IDIM * 2);
  __bf16* Wk1b  = (__bf16*)carve((size_t)D_MODEL * IDIM * 2);
  __bf16* Wv1b  = (__bf16*)carve((size_t)D_MODEL * IDIM * 2);
  __bf16* Wo1b  = (__bf16*)carve((size_t)IDIM * D_MODEL * 2);
  __bf16* Wq2b  = (__bf16*)carve((size_t)D_MODEL * IDIM * 2);
  __bf16* Wk2b  = (__bf16*)carve((size_t)DCTX * IDIM * 2);
  __bf16* Wv2b  = (__bf16*)carve((size_t)DCTX * IDIM * 2);
  __bf16* Wo2b  = (__bf16*)carve((size_t)IDIM * D_MODEL * 2);
  __bf16* Wff1b = (__bf16*)carve((size_t)D_MODEL * 2 * FFDIM * 2);
  __bf16* Wff2b = (__bf16*)carve((size_t)FFDIM * D_MODEL * 2);
  __bf16* ctxb  = (__bf16*)carve((size_t)BATCH * MPAD * DCTX * 2);
  __bf16* xn    = (__bf16*)carve((size_t)TOKENS * D_MODEL * 2);
  __bf16* qb    = (__bf16*)carve((size_t)TOKENS * IDIM * 2);
  __bf16* kb    = (__bf16*)carve((size_t)TOKENS * IDIM * 2);
  __bf16* vb    = (__bf16*)carve((size_t)TOKENS * IDIM * 2);
  __bf16* vTb   = (__bf16*)carve((size_t)TOKENS * IDIM * 2);
  __bf16* ao    = (__bf16*)carve((size_t)TOKENS * IDIM * 2);
  float*  xr    = (float*)carve((size_t)TOKENS * D_MODEL * 4);
  __bf16* hg    = (__bf16*)carve((size_t)TOKENS * 2 * FFDIM * 2);
  __bf16* ffh   = (__bf16*)carve((size_t)TOKENS * FFDIM * 2);

  auto cvt = [&](const float* src, __bf16* dst, size_t n) {
    int blocks = (int)((n + 255) / 256); if (blocks > 4096) blocks = 4096;
    cvt_bf16_kernel<<<blocks, 256, 0, stream>>>(src, dst, n);
  };
  cvt(Wq1,  Wq1b,  (size_t)D_MODEL * IDIM);
  cvt(Wk1,  Wk1b,  (size_t)D_MODEL * IDIM);
  cvt(Wv1,  Wv1b,  (size_t)D_MODEL * IDIM);
  cvt(Wo1,  Wo1b,  (size_t)IDIM * D_MODEL);
  cvt(Wq2,  Wq2b,  (size_t)D_MODEL * IDIM);
  cvt(Wk2,  Wk2b,  (size_t)DCTX * IDIM);
  cvt(Wv2,  Wv2b,  (size_t)DCTX * IDIM);
  cvt(Wo2,  Wo2b,  (size_t)IDIM * D_MODEL);
  cvt(Wff1, Wff1b, (size_t)D_MODEL * 2 * FFDIM);
  cvt(Wff2, Wff2b, (size_t)FFDIM * D_MODEL);

  const dim3 blk256(256);
  const dim3 gN1024(IDIM / 128, TOKENS / 64);          // (8, 64)
  const dim3 gFF1(2 * FFDIM / 128, TOKENS / 64);       // (64, 64)
  const dim3 gKV2(IDIM / 128, (BATCH * MPAD) / 64);    // (8, 3)
  const dim3 gFF2(D_MODEL / 128, TOKENS / 64);         // (8, 64)
  const dim3 gAttn(N_TOK / 16, NHEAD, BATCH);          // (128, 16, 2)

  // ---- Block 1: self-attention -------------------------------------------
  ln_kernel<<<TOKENS, blk256, 0, stream>>>(x, ln1g, ln1b, xn);
  gemm_bf16_wmma<0><<<gN1024, blk256, 0, stream>>>(xn, Wq1b, nullptr, nullptr, qb, TOKENS, IDIM, D_MODEL);
  gemm_bf16_wmma<0><<<gN1024, blk256, 0, stream>>>(xn, Wk1b, nullptr, nullptr, kb, TOKENS, IDIM, D_MODEL);
  gemm_bf16_wmma<0><<<gN1024, blk256, 0, stream>>>(xn, Wv1b, nullptr, nullptr, vb, TOKENS, IDIM, D_MODEL);
  vtrans_kernel<<<4096, blk256, 0, stream>>>(vb, vTb, N_TOK);
  attn_kernel<<<gAttn, dim3(32), 0, stream>>>(qb, kb, vTb, ao, N_TOK, N_TOK, N_TOK, SCALE_F);
  gemm_bf16_wmma<2><<<gFF2, blk256, 0, stream>>>(ao, Wo1b, bo1, x, xr, TOKENS, D_MODEL, IDIM);

  // ---- Block 2: cross-attention ------------------------------------------
  ln_kernel<<<TOKENS, blk256, 0, stream>>>(xr, ln2g, ln2b, xn);
  ctxpad_kernel<<<576, blk256, 0, stream>>>(ctx, ctxb);
  gemm_bf16_wmma<0><<<gN1024, blk256, 0, stream>>>(xn, Wq2b, nullptr, nullptr, qb, TOKENS, IDIM, D_MODEL);
  gemm_bf16_wmma<0><<<gKV2,  blk256, 0, stream>>>(ctxb, Wk2b, nullptr, nullptr, kb, BATCH * MPAD, IDIM, DCTX);
  gemm_bf16_wmma<0><<<gKV2,  blk256, 0, stream>>>(ctxb, Wv2b, nullptr, nullptr, vb, BATCH * MPAD, IDIM, DCTX);
  vtrans_kernel<<<768, blk256, 0, stream>>>(vb, vTb, MPAD);
  attn_kernel<<<gAttn, dim3(32), 0, stream>>>(qb, kb, vTb, ao, N_TOK, MCTX, MPAD, SCALE_F);
  gemm_bf16_wmma<2><<<gFF2, blk256, 0, stream>>>(ao, Wo2b, bo2, xr, xr, TOKENS, D_MODEL, IDIM);

  // ---- Block 3: GEGLU FFN -------------------------------------------------
  ln_kernel<<<TOKENS, blk256, 0, stream>>>(xr, ln3g, ln3b, xn);
  gemm_bf16_wmma<1><<<gFF1, blk256, 0, stream>>>(xn, Wff1b, bff1, nullptr, hg, TOKENS, 2 * FFDIM, D_MODEL);
  geglu_kernel<<<4096, blk256, 0, stream>>>(hg, ffh);
  gemm_bf16_wmma<2><<<gFF2, blk256, 0, stream>>>(ffh, Wff2b, bff2, xr, d_out, TOKENS, D_MODEL, FFDIM);
}
